// DeepCrossing_47614007443581
// MI455X (gfx1250) — matrix operand order repaired and checked by reading, not compiled, not run on hardware
//
#include <hip/hip_runtime.h>
#include <hip/hip_bf16.h>
#include <math.h>

typedef __attribute__((ext_vector_type(16))) __bf16 v16bf;
typedef __attribute__((ext_vector_type(8)))  __bf16 v8bf;
typedef __attribute__((ext_vector_type(8)))  float  v8f;

#define NBATCH 16384
#define NF     26
#define ED     16
#define NV     100001      // V+1 rows per table
#define HID    256
#define NDENSE 13
#define STACK  429
#define KPAD   448         // 14 * 32
#define BT     128         // batch rows per workgroup
#define XS     456         // Xs LDS row stride (bf16 elems)
#define HS     264         // Hs LDS row stride
#define WTS    40          // weight tile per-column stride (32 k + 8 pad)

#define KC1    14          // K chunks GEMM1 (448/32), even
#define KC2    8           // K chunks GEMM2 (256/32), even
#define WT1_ELEMS (3 * KC1 * 256 * WTS)   // 430080 bf16
#define WT2_ELEMS (3 * KC2 * 512 * WTS)   // 491520 bf16
#define TILE_ELEMS (128 * WTS)            // 5120 bf16 = 10240 B per staged chunk

__device__ __forceinline__ __bf16 f2bf(float f) {
  unsigned u = __builtin_bit_cast(unsigned, f);
  u += 0x7FFFu + ((u >> 16) & 1u);       // round-to-nearest-even
  unsigned short h = (unsigned short)(u >> 16);
  return __builtin_bit_cast(__bf16, h);
}
__device__ __forceinline__ float bf2f(__bf16 b) {
  unsigned short h = __builtin_bit_cast(unsigned short, b);
  return __builtin_bit_cast(float, ((unsigned)h) << 16);
}
__device__ __forceinline__ v16bf cat8(v8bf lo, v8bf hi) {
  return __builtin_shufflevector(lo, hi, 0,1,2,3,4,5,6,7,8,9,10,11,12,13,14,15);
}
// A fragment 16x32 bf16: lane<16 holds K 0..7 & 16..23, lane>=16 holds 8..15 & 24..31
__device__ __forceinline__ v16bf load_a(const __bf16* base /* &A[m*ld + kb] */, int lh) {
  v8bf lo = *(const v8bf*)(base + lh * 8);
  v8bf hi = *(const v8bf*)(base + 16 + lh * 8);
  return cat8(lo, hi);
}
// B fragment 32x16 bf16 from transposed staging: lane group lh holds K = lh*8 .. +7 & 16+lh*8 .. +7
__device__ __forceinline__ v16bf load_b(const __bf16* WbT, int nloc, int lh) {
  const __bf16* p = WbT + nloc * WTS + lh * 8;
  return cat8(*(const v8bf*)p, *(const v8bf*)(p + 16));
}
// issue async DMA of one 10240B weight tile into LDS (no wait) — whole 256-thread block
__device__ __forceinline__ void async_issue_tile(const __bf16* src, unsigned ldsOff, int tid) {
  unsigned long long gb = (unsigned long long)(size_t)src;
  #pragma unroll
  for (int i = 0; i < 5; ++i) {          // 5 * 256 lanes * 8B = 10240 B
    unsigned lds = ldsOff + i * 2048 + tid * 8;
    unsigned long long ga = gb + i * 2048 + tid * 8;
    asm volatile("global_load_async_to_lds_b64 %0, %1, off"
                 :: "v"(lds), "v"(ga) : "memory");
  }
}
__device__ __forceinline__ void async_wait0() {
  asm volatile("s_wait_asynccnt 0" ::: "memory");
}

// ---- one-time weight conversion: f32 -> bf16, transposed padded tiles in ws ----
__global__ void __launch_bounds__(256)
prep_weights(const float* __restrict__ w1s, const float* __restrict__ w2s,
             __bf16* __restrict__ wt1, __bf16* __restrict__ wt2) {
  int i = blockIdx.x * 256 + threadIdx.x;
  if (i < WT1_ELEMS) {
    int k   = i % WTS;
    int rem = i / WTS;            // (blk*KC1 + kc)*256 + n
    int n   = rem & 255;
    int ck  = rem >> 8;
    int blk = ck / KC1, kc = ck % KC1;
    int gk  = kc * 32 + k;
    float w = (k < 32 && gk < STACK) ? w1s[(blk * STACK + gk) * HID + n] : 0.f;
    wt1[i] = f2bf(w);
  } else if (i < WT1_ELEMS + WT2_ELEMS) {
    int ii  = i - WT1_ELEMS;
    int k   = ii % WTS;
    int rem = ii / WTS;           // (blk*KC2 + kc)*512 + n
    int n   = rem & 511;
    int ck  = rem >> 9;
    int blk = ck / KC2, kc = ck % KC2;
    int gk  = kc * 32 + k;
    float w = (k < 32 && n < STACK) ? w2s[(blk * HID + gk) * STACK + n] : 0.f;
    wt2[ii] = f2bf(w);
  }
}

__global__ void __launch_bounds__(256, 1)
deepcross_fused(const float* __restrict__ dense,
                const int*   __restrict__ sparse,
                const int*   __restrict__ label,
                const float* __restrict__ emb,
                const __bf16* __restrict__ wt1, const float* __restrict__ b1s,
                const __bf16* __restrict__ wt2, const float* __restrict__ b2s,
                const float* __restrict__ wlast, const float* __restrict__ blast,
                float* __restrict__ out)
{
  extern __shared__ __align__(32) char smem[];
  __bf16* Xs   = (__bf16*)smem;          // [BT][XS]
  __bf16* Hs   = Xs + BT * XS;           // [BT][HS]
  __bf16* WbT0 = Hs + BT * HS;           // [128][WTS] weight chunk, buffer 0
  __bf16* WbT1 = WbT0 + TILE_ELEMS;      // buffer 1

  const int tid  = threadIdx.x;
  const int wave = tid >> 5;
  const int lane = tid & 31;
  const int ln   = lane & 15;
  const int lh   = lane >> 4;
  const int m0   = wave * 16;            // each wave owns 16 batch rows
  const int row0 = blockIdx.x * BT;
  const unsigned wb0Off = (unsigned)(size_t)WbT0;   // low 32 bits = LDS offset
  const unsigned wb1Off = (unsigned)(size_t)WbT1;

  // ---- gather: embeddings + dense -> Xs (bf16), zero-pad cols [429,448) ----
  for (int e = tid; e < BT * KPAD; e += 256) {
    int r = e / KPAD;
    int c = e - r * KPAD;
    float v = 0.f;
    if (c < NF * ED) {
      int f = c >> 4, d = c & 15;
      int idx = sparse[(row0 + r) * NF + f];
      v = emb[(f * NV + idx) * ED + d];
    } else if (c < STACK) {
      v = dense[(row0 + r) * NDENSE + (c - NF * ED)];
    }
    Xs[r * XS + c] = f2bf(v);
  }

  #pragma unroll 1
  for (int blk = 0; blk < 3; ++blk) {
    const float* B1 = b1s + blk * HID;
    const float* B2 = b2s + blk * STACK;

    // ---------- GEMM1: Hs = Xs @ W1 + b1   (M=128, N=256, K=448) ----------
    #pragma unroll 1
    for (int pass = 0; pass < 2; ++pass) {
      const int cb = pass * 128;
      const __bf16* wbase = wt1 + ((size_t)(blk * KC1) * 256 + cb) * WTS;  // + kc*256*WTS
      const __bf16* Arow  = &Xs[(m0 + ln) * XS];
      v8f acc[8] = {};

      auto compute_chunk = [&](const __bf16* cur, int kb) {
        v16bf a = load_a(Arow + kb, lh);
        #pragma unroll
        for (int t = 0; t < 8; ++t) {
          v16bf b = load_b(cur, t * 16 + ln, lh);
          acc[t] = __builtin_amdgcn_wmma_f32_16x16x32_bf16(
              false, a, false, b, (short)0, acc[t], false, false);
        }
      };

      __syncthreads();                                 // WbT buffers free
      async_issue_tile(wbase, wb0Off, tid);            // stage kc=0 -> buf0
      #pragma unroll 1
      for (int kc = 0; kc < KC1; kc += 2) {
        // even chunk: compute buf0, DMA kc+1 -> buf1
        async_wait0();
        __syncthreads();
        async_issue_tile(wbase + (size_t)(kc + 1) * 256 * WTS, wb1Off, tid);
        compute_chunk(WbT0, kc * 32);
        // odd chunk: compute buf1, DMA kc+2 -> buf0
        async_wait0();
        __syncthreads();
        if (kc + 2 < KC1)
          async_issue_tile(wbase + (size_t)(kc + 2) * 256 * WTS, wb0Off, tid);
        compute_chunk(WbT1, (kc + 1) * 32);
      }
      #pragma unroll
      for (int t = 0; t < 8; ++t) {
        int n = cb + t * 16 + ln;
        float bias = B1[n];
        #pragma unroll
        for (int v = 0; v < 8; ++v)
          Hs[(m0 + lh * 8 + v) * HS + n] = f2bf(acc[t][v] + bias);
      }
    }

    // ------ GEMM2 + residual: Xs = relu(Hs @ W2 + b2 + Xs)  (N=448, K=256) ------
    #pragma unroll 1
    for (int pass = 0; pass < 4; ++pass) {
      const int cb = pass * 128;
      const __bf16* wbase = wt2 + ((size_t)(blk * KC2) * 512 + cb) * WTS;  // + kc*512*WTS
      const __bf16* Arow  = &Hs[(m0 + ln) * HS];
      v8f acc[8] = {};

      auto compute_chunk = [&](const __bf16* cur, int kb) {
        v16bf a = load_a(Arow + kb, lh);
        #pragma unroll
        for (int t = 0; t < 8; ++t) {
          v16bf b = load_b(cur, t * 16 + ln, lh);
          acc[t] = __builtin_amdgcn_wmma_f32_16x16x32_bf16(
              false, a, false, b, (short)0, acc[t], false, false);
        }
      };

      __syncthreads();
      async_issue_tile(wbase, wb0Off, tid);
      #pragma unroll 1
      for (int kc = 0; kc < KC2; kc += 2) {
        async_wait0();
        __syncthreads();
        async_issue_tile(wbase + (size_t)(kc + 1) * 512 * WTS, wb1Off, tid);
        compute_chunk(WbT0, kc * 32);
        async_wait0();
        __syncthreads();
        if (kc + 2 < KC2)
          async_issue_tile(wbase + (size_t)(kc + 2) * 512 * WTS, wb0Off, tid);
        compute_chunk(WbT1, (kc + 1) * 32);
      }
      #pragma unroll
      for (int t = 0; t < 8; ++t) {
        int n = cb + t * 16 + ln;
        if (n < KPAD) {
          float bias = (n < STACK) ? B2[n] : 0.f;
          #pragma unroll
          for (int v = 0; v < 8; ++v) {
            int o = (m0 + lh * 8 + v) * XS + n;
            float y = acc[t][v] + bias + bf2f(Xs[o]);
            Xs[o] = f2bf(y > 0.f ? y : 0.f);
          }
        }
      }
    }
  }

  __syncthreads();
  // ---- head: sigmoid(Xs @ w_last + b_last); 2 threads per row ----
  {
    int r = tid >> 1, half = tid & 1;
    float s = 0.f;
    for (int c = half; c < STACK; c += 2)
      s += bf2f(Xs[r * XS + c]) * wlast[c];
    s += __shfl_xor(s, 1, 32);
    if (half == 0) {
      float z = s + blast[0];
      out[row0 + r] = 1.f / (1.f + expf(-z));
    }
    if (tid < BT) out[NBATCH + row0 + tid] = (float)label[row0 + tid];
  }
}

extern "C" void kernel_launch(void* const* d_in, const int* in_sizes, int n_in,
                              void* d_out, int out_size, void* d_ws, size_t ws_size,
                              hipStream_t stream) {
  const float* dense  = (const float*)d_in[0];
  const int*   sparse = (const int*)d_in[1];
  const int*   label  = (const int*)d_in[2];
  const float* emb    = (const float*)d_in[3];
  const float* w1s    = (const float*)d_in[4];
  const float* b1s    = (const float*)d_in[5];
  const float* w2s    = (const float*)d_in[6];
  const float* b2s    = (const float*)d_in[7];
  const float* wlast  = (const float*)d_in[8];
  const float* blast  = (const float*)d_in[9];
  float* out = (float*)d_out;

  __bf16* wt1 = (__bf16*)d_ws;
  __bf16* wt2 = wt1 + WT1_ELEMS;

  int prep_total = WT1_ELEMS + WT2_ELEMS;
  hipLaunchKernelGGL(prep_weights, dim3((prep_total + 255) / 256), dim3(256), 0, stream,
                     w1s, w2s, wt1, wt2);

  size_t shmem = (size_t)(BT * XS + BT * HS + 2 * TILE_ELEMS) * 2;  // ~205 KB LDS
  hipLaunchKernelGGL(deepcross_fused, dim3(NBATCH / BT), dim3(256), shmem, stream,
                     dense, sparse, label, emb, wt1, b1s, wt2, b2s, wlast, blast, out);
}
